// Attention_16801912062679
// MI455X (gfx1250) — matrix operand order repaired and checked by reading, not compiled, not run on hardware
//
#include <hip/hip_runtime.h>

// ---- CDNA5 (gfx1250) wave32 WMMA fused attention ----
// B=8, T=2048, D_MODEL=1024, HEAD_DIM=128, causal, f32 in/out, bf16 compute.

typedef __attribute__((ext_vector_type(16))) __bf16 v16bf;
typedef __attribute__((ext_vector_type(8)))  __bf16 v8bf;
typedef __attribute__((ext_vector_type(8)))  float  v8f;
typedef int v4i32 __attribute__((vector_size(16)));   // matches builtin param type

union V16U { v16bf v; v8bf h[2]; };

static constexpr int Bn = 8;
static constexpr int Tn = 2048;
static constexpr int Dm = 1024;
static constexpr int HD = 128;
static constexpr float QSCALE = 0.08838834764831845f; // 1/sqrt(128)

__device__ __forceinline__ v8f wmma_bf16(const v16bf& a, const v16bf& b, const v8f& c) {
    return __builtin_amdgcn_wmma_f32_16x16x32_bf16(false, a, false, b, (short)0, c, false, false);
}

// Async global -> LDS 16-byte copy (ASYNCcnt-tracked); sync fallback otherwise.
__device__ __forceinline__ void async_copy16(const __bf16* g, __bf16* l) {
#if __has_builtin(__builtin_amdgcn_global_load_async_to_lds_b128)
    __builtin_amdgcn_global_load_async_to_lds_b128(
        (__attribute__((address_space(1))) v4i32*)g,
        (__attribute__((address_space(3))) v4i32*)l,
        0, 0);
#else
    *(v8bf*)l = *(const v8bf*)g;
#endif
}

__device__ __forceinline__ void wait_async() {
#if __has_builtin(__builtin_amdgcn_s_wait_asynccnt)
    __builtin_amdgcn_s_wait_asynccnt(0);
#else
    asm volatile("s_wait_asynccnt 0" ::: "memory");
#endif
}

// ---------------- Kernel 0: one-time f32 -> bf16 weight conversion ----------
__global__ void __launch_bounds__(256)
cvt_w_kernel(const float* __restrict__ src, __bf16* __restrict__ dst, int n8) {
    const int i = blockIdx.x * blockDim.x + threadIdx.x;
    if (i < n8) {
        const v8f f = *(const v8f*)(src + (size_t)i * 8);
        v8bf o;
#pragma unroll
        for (int e = 0; e < 8; ++e) o[e] = (__bf16)f[e];
        *(v8bf*)(dst + (size_t)i * 8) = o;
    }
}

// ---------------- Kernel 1: QKV projection ----------------
// grid = (B*T)/64 blocks, 128 threads (4 waves). Wave handles 16 tokens.
// y = x @ W^T with bf16 W[128,1024]; Q scaled by 1/sqrt(HD); V stored transposed.
__global__ void __launch_bounds__(128)
qkv_proj_kernel(const float* __restrict__ x,
                const __bf16* __restrict__ Wqb,
                const __bf16* __restrict__ Wkb,
                const __bf16* __restrict__ Wvb,
                __bf16* __restrict__ Q,
                __bf16* __restrict__ K,
                __bf16* __restrict__ Vt) {
    __shared__ __bf16 stage[4][16][HD];  // 16 KB: per-wave 16x128 bf16 tile

    const int wave = threadIdx.x >> 5;
    const int lane = threadIdx.x & 31;
    const int l15  = lane & 15;
    const int half = lane >> 4;
    const int rowbase = blockIdx.x * 64 + wave * 16;

    const v8f z = {0.f, 0.f, 0.f, 0.f, 0.f, 0.f, 0.f, 0.f};

    // All 8 B operands of a kc step preloaded into distinct registers so the
    // scheduler can keep loads in flight instead of s_wait_loadcnt 0 per wmma.
    auto compute = [&](const __bf16* __restrict__ Wb, v8f (&acc)[8]) {
#pragma unroll
        for (int h = 0; h < 8; ++h) acc[h] = z;
        for (int kc = 0; kc < Dm / 32; ++kc) {
            const float* xr = x + (size_t)(rowbase + l15) * Dm + 32 * kc;
            const v8f f0 = *(const v8f*)(xr + 16 * half);
            const v8f f1 = *(const v8f*)(xr + 16 * half + 8);
            v16bf bm[8];
#pragma unroll
            for (int nh = 0; nh < 8; ++nh)
                bm[nh] = *(const v16bf*)(
                    Wb + (size_t)(nh * 16 + l15) * Dm + 32 * kc + 16 * half);
            V16U a;
#pragma unroll
            for (int e = 0; e < 8; ++e) {
                a.v[e]     = (__bf16)f0[e];
                a.v[e + 8] = (__bf16)f1[e];
            }
#pragma unroll
            for (int nh = 0; nh < 8; ++nh)
                acc[nh] = wmma_bf16(a.v, bm[nh], acc[nh]);
        }
    };

    auto to_stage = [&](v8f (&acc)[8], float sc) {
#pragma unroll
        for (int h = 0; h < 8; ++h)
#pragma unroll
            for (int r = 0; r < 8; ++r)
                stage[wave][r + 8 * half][h * 16 + l15] = (__bf16)(acc[h][r] * sc);
    };

    auto flush_rowmajor = [&](__bf16* __restrict__ dst) {
        asm volatile("s_wait_dscnt 0" ::: "memory");
        const int row = lane >> 1;
        const int seg = lane & 1;
        const __bf16* src = &stage[wave][row][seg * 64];
        __bf16* drow = dst + (size_t)(rowbase + row) * HD + seg * 64;
#pragma unroll
        for (int i = 0; i < 8; ++i)
            *(v8bf*)(drow + i * 8) = *(const v8bf*)(src + i * 8);
    };

    auto flush_transposed = [&](__bf16* __restrict__ dst) {
        asm volatile("s_wait_dscnt 0" ::: "memory");
        const int bidx = rowbase / Tn;
        const int tb   = rowbase % Tn;
#pragma unroll
        for (int j = 0; j < 4; ++j) {
            const int hd = j * 32 + lane;
            v8bf lo, hi;
#pragma unroll
            for (int t = 0; t < 8; ++t) {
                lo[t] = stage[wave][t][hd];
                hi[t] = stage[wave][t + 8][hd];
            }
            __bf16* drow = dst + ((size_t)bidx * HD + hd) * Tn + tb;
            *(v8bf*)(drow)     = lo;
            *(v8bf*)(drow + 8) = hi;
        }
    };

    v8f acc[8];

    compute(Wqb, acc);
    to_stage(acc, QSCALE);
    flush_rowmajor(Q);

    compute(Wkb, acc);
    to_stage(acc, 1.0f);
    flush_rowmajor(K);

    compute(Wvb, acc);
    to_stage(acc, 1.0f);
    flush_transposed(Vt);
}

// ---------------- Kernel 2: causal flash attention ----------------
// grid = B * (T/64) blocks, 128 threads (4 waves). Wave owns 16 query rows.
// K/V tiles are loaded cooperatively into double-buffered LDS via the async
// global->LDS path; next tile's fill overlaps current tile's WMMAs.
__global__ void __launch_bounds__(128)
flash_attn_kernel(const __bf16* __restrict__ Q,
                  const __bf16* __restrict__ K,
                  const __bf16* __restrict__ Vt,
                  float* __restrict__ out) {
    // Padded rows: K row 136 (272 B, +4 banks/row), V row 40 (80 B, +20 banks)
    // -> conflict-free 16-lane column reads of the B operands.
    __shared__ __bf16 ktile[2][32][136];   // ~17.4 KB
    __shared__ __bf16 vtile[2][128][40];   // 20 KB
    __shared__ __bf16 pbuf[4][16][32];     // per-wave P staging (C/D -> A)

    const int b    = blockIdx.x >> 5;
    const int qt   = blockIdx.x & 31;
    const int wave = threadIdx.x >> 5;
    const int lane = threadIdx.x & 31;
    const int tid  = threadIdx.x;
    const int l15  = lane & 15;
    const int half = lane >> 4;
    const int q0   = qt * 64 + wave * 16;   // wave's first query row
    const int qblk = qt * 64;               // block's first query row

    const __bf16* Qb  = Q  + (size_t)b * Tn * HD;
    const __bf16* Kb  = K  + (size_t)b * Tn * HD;
    const __bf16* Vtb = Vt + (size_t)b * HD * Tn;

    // Cooperative fill of one 32-key K/V tile pair (8 KB + 8 KB, 16 B/chunk).
    auto issue_tile = [&](int kbase, int buf) {
#pragma unroll
        for (int i = 0; i < 4; ++i) {           // K: 512 chunks / 128 threads
            const int c   = tid + i * 128;
            const int row = c >> 4, seg = c & 15;
            async_copy16(Kb + (size_t)(kbase + row) * HD + seg * 8,
                         &ktile[buf][row][seg * 8]);
        }
#pragma unroll
        for (int i = 0; i < 4; ++i) {           // V: 512 chunks / 128 threads
            const int c   = tid + i * 128;
            const int row = c >> 2, seg = c & 3;
            async_copy16(Vtb + (size_t)row * Tn + kbase + seg * 8,
                         &vtile[buf][row][seg * 8]);
        }
    };

    // Q A-operands resident for the whole loop (4 chunks of K=32).
    V16U qa[4];
    const __bf16* qrow = Qb + (size_t)(q0 + l15) * HD;
#pragma unroll
    for (int c = 0; c < 4; ++c) {
        qa[c].h[0] = *(const v8bf*)(qrow + 32 * c + 16 * half);
        qa[c].h[1] = *(const v8bf*)(qrow + 32 * c + 16 * half + 8);
    }

    const v8f z = {0.f, 0.f, 0.f, 0.f, 0.f, 0.f, 0.f, 0.f};
    v8f o[8];
#pragma unroll
    for (int h = 0; h < 8; ++h) o[h] = z;
    float m[8], l[8];
#pragma unroll
    for (int r = 0; r < 8; ++r) { m[r] = -1e30f; l[r] = 0.f; }

    const int nkt = (qblk + 64 + 31) >> 5;  // block-level tile count
    issue_tile(0, 0);

    for (int kt = 0; kt < nkt; ++kt) {
        const int kbase = kt * 32;
        const int buf   = kt & 1;

        wait_async();
        __syncthreads();                     // tile kt resident for all waves
        if (kt + 1 < nkt) issue_tile(kbase + 32, buf ^ 1);

        if (kbase <= q0 + 15) {              // uniform: wave has live keys here
            // S tile: 16 q-rows x 32 keys, contract over HD=128 (LDS B ops).
            v8f s0 = z, s1 = z;
#pragma unroll
            for (int c = 0; c < 4; ++c) {
                const v16bf kb0 = *(const v16bf*)&ktile[buf][l15][32 * c + 16 * half];
                s0 = wmma_bf16(qa[c].v, kb0, s0);
                const v16bf kb1 = *(const v16bf*)&ktile[buf][16 + l15][32 * c + 16 * half];
                s1 = wmma_bf16(qa[c].v, kb1, s1);
            }

            // Causal mask (softmax scale already folded into Q).
            const int key0 = kbase + l15;
            const int key1 = key0 + 16;
#pragma unroll
            for (int r = 0; r < 8; ++r) {
                const int qr = q0 + r + 8 * half;
                if (key0 > qr) s0[r] = -1e30f;
                if (key1 > qr) s1[r] = -1e30f;
            }

            // Online softmax: per-row stats over each 16-lane half-wave.
            float alpha[8];
#pragma unroll
            for (int r = 0; r < 8; ++r) {
                float rmax = fmaxf(s0[r], s1[r]);
#pragma unroll
                for (int off = 1; off < 16; off <<= 1)
                    rmax = fmaxf(rmax, __shfl_xor(rmax, off, 32));
                const float mnew = fmaxf(m[r], rmax);
                alpha[r] = __expf(m[r] - mnew);
                m[r] = mnew;
                s0[r] = __expf(s0[r] - mnew);
                s1[r] = __expf(s1[r] - mnew);
                float rs = s0[r] + s1[r];
#pragma unroll
                for (int off = 1; off < 16; off <<= 1)
                    rs += __shfl_xor(rs, off, 32);
                l[r] = l[r] * alpha[r] + rs;
            }
#pragma unroll
            for (int h = 0; h < 8; ++h)
#pragma unroll
                for (int r = 0; r < 8; ++r) o[h][r] *= alpha[r];

            // P (C/D layout) -> LDS -> P (A layout, 16x32 bf16).
            __bf16* pw = &pbuf[wave][0][0];
#pragma unroll
            for (int r = 0; r < 8; ++r) {
                pw[(r + 8 * half) * 32 + l15]      = (__bf16)s0[r];
                pw[(r + 8 * half) * 32 + 16 + l15] = (__bf16)s1[r];
            }
            asm volatile("s_wait_dscnt 0" ::: "memory");
            V16U pa;
            const __bf16* prow = pw + l15 * 32;
            pa.h[0] = *(const v8bf*)(prow + 16 * half);
            pa.h[1] = *(const v8bf*)(prow + 16 * half + 8);

            // O += P @ V (V B-operand = 16 contiguous keys from LDS tile).
#pragma unroll
            for (int h = 0; h < 8; ++h) {
                const v16bf vb = *(const v16bf*)&vtile[buf][h * 16 + l15][16 * half];
                o[h] = wmma_bf16(pa.v, vb, o[h]);
            }
        }
    }

    // Normalize and store f32 output [B, T, HD] (nontemporal: write-once).
#pragma unroll
    for (int r = 0; r < 8; ++r) l[r] = 1.0f / l[r];
#pragma unroll
    for (int h = 0; h < 8; ++h)
#pragma unroll
        for (int r = 0; r < 8; ++r) {
            const int row = q0 + r + 8 * half;
            __builtin_nontemporal_store(
                o[h][r] * l[r],
                &out[((size_t)b * Tn + row) * HD + h * 16 + l15]);
        }
}

extern "C" void kernel_launch(void* const* d_in, const int* in_sizes, int n_in,
                              void* d_out, int out_size, void* d_ws, size_t ws_size,
                              hipStream_t stream) {
    (void)in_sizes; (void)n_in; (void)out_size; (void)ws_size;
    const float* x  = (const float*)d_in[0];
    const float* Wq = (const float*)d_in[1];
    const float* Wk = (const float*)d_in[2];
    const float* Wv = (const float*)d_in[3];
    float* out = (float*)d_out;

    __bf16* Qws = (__bf16*)d_ws;
    __bf16* Kws = Qws + (size_t)Bn * Tn * HD;   // +4 MB
    __bf16* Vtw = Kws + (size_t)Bn * Tn * HD;   // +4 MB
    __bf16* Wqb = Vtw + (size_t)Bn * Tn * HD;   // +4 MB
    __bf16* Wkb = Wqb + (size_t)HD * Dm;        // +256 KB
    __bf16* Wvb = Wkb + (size_t)HD * Dm;

    const int wn8 = (HD * Dm) / 8;              // 16384 vec8 elements per weight
    const int cvtBlocks = (wn8 + 255) / 256;    // 64
    cvt_w_kernel<<<cvtBlocks, 256, 0, stream>>>(Wq, Wqb, wn8);
    cvt_w_kernel<<<cvtBlocks, 256, 0, stream>>>(Wk, Wkb, wn8);
    cvt_w_kernel<<<cvtBlocks, 256, 0, stream>>>(Wv, Wvb, wn8);

    const int blocks = (Bn * Tn) / 64;  // 256
    qkv_proj_kernel<<<blocks, 128, 0, stream>>>(x, Wqb, Wkb, Wvb, Qws, Kws, Vtw);
    flash_attn_kernel<<<blocks, 128, 0, stream>>>(Qws, Kws, Vtw, out);
}
